// ImprovedRouteOptimizationGNN_61220463837373
// MI455X (gfx1250) — compile-verified, matrix operand order checked by reading, not run on hardware
//
#include <hip/hip_runtime.h>
#include <hip/hip_bf16.h>
#include <math.h>
#include <stddef.h>
#include <stdint.h>

#define N_NODES 10000
#define N_EDGES 160000
#define F_INP   8
#define HID     128
#define HEADS   8
#define LAYERS  4
#define EDIM    4
#define QKV     (HEADS*HID)   /* 1024 */
#define EPSV    1e-5f

typedef __attribute__((ext_vector_type(16))) __bf16 v16bf;
typedef __attribute__((ext_vector_type(8)))  __bf16 bf16x8;
typedef __attribute__((ext_vector_type(4)))  __bf16 bf16x4;
typedef __attribute__((ext_vector_type(8)))  float  v8f;

// ---------------------------------------------------------------- helpers
__device__ __forceinline__ void atomicMaxFloat(float* addr, float val) {
  if (val >= 0.0f) atomicMax((int*)addr, __float_as_int(val));
  else             atomicMin((unsigned int*)addr, (unsigned int)__float_as_int(val));
}

__global__ void fill_kernel(float* p, float val, int n) {
  int i = blockIdx.x * blockDim.x + threadIdx.x;
  if (i < n) p[i] = val;
}

__global__ void scale2_kernel(float* p, int n) {
  int i = blockIdx.x * blockDim.x + threadIdx.x;
  if (i < n) p[i] *= 2.0f;
}

__global__ void relu_kernel(float* p, int n) {
  int i = blockIdx.x * blockDim.x + threadIdx.x;
  if (i < n) p[i] = fmaxf(p[i], 0.0f);
}

__global__ void add2_kernel(float* out, const float* a, const float* b, int n) {
  int i = blockIdx.x * blockDim.x + threadIdx.x;
  if (i < n) out[i] = a[i] + b[i];
}

// =================================================================
// bf16 WMMA GEMM, LDS-staged.  Block = 256 threads = 8 waves.
// Block tile: 128 (M) x 64 (N).  Wave w -> rows [w*16, w*16+16), all 64 cols
// (4 accumulators, A fragment reused 4x).  B tile staged via CDNA5
// GLOBAL_LOAD_ASYNC_TO_LDS_B128 (ASYNCcnt path), A tile via coalesced b128
// loads + v_cvt_pk_bf16 + ds_store_b64.  Fragments = 2x ds_load_b128 each.
// Requires: K % 32 == 0, NC % 64 == 0.  M arbitrary (clamped/guarded).
// =================================================================
struct __align__(16) GemmSmem {
  float  braw[32 * 64];    // raw f32 B tile (async landing zone), 8 KB
  __bf16 a[128 * 32];      // A tile, K-contiguous bf16, 8 KB
  __bf16 bt[64 * 32];      // B tile transposed [n][k] bf16, 4 KB
};

__global__ __launch_bounds__(256)
void wmma_gemm_bias(const float* __restrict__ A, const float* __restrict__ B,
                    const float* __restrict__ bias, float* __restrict__ C,
                    int M, int K, int NC) {
  __shared__ GemmSmem sm;
  const int tid  = threadIdx.x;
  const int wave = tid >> 5;
  const int lane = tid & 31;
  const int lm   = lane & 15;
  const int khi  = lane >> 4;          // K-half select per ISA bf16 layout
  const int row0 = blockIdx.x * 128;
  const int col0 = blockIdx.y * 64;

  v8f acc[4] = {v8f{}, v8f{}, v8f{}, v8f{}};

  for (int kb = 0; kb < K; kb += 32) {
    // ---- B tile: async global -> LDS (raw f32, 32x64), 2x b128 per thread
#pragma unroll
    for (int u = 0; u < 2; ++u) {
      int flat = u * 1024 + tid * 4;
      int kk = flat >> 6, nn = flat & 63;
      const float* gp = B + (size_t)(kb + kk) * NC + (col0 + nn);
      unsigned laddr = (unsigned)(offsetof(GemmSmem, braw) + (size_t)flat * 4);
      asm volatile("global_load_async_to_lds_b128 %0, %1, off"
                   :: "v"(laddr), "v"((unsigned long long)(uintptr_t)gp)
                   : "memory");
    }
    // ---- A tile: coalesced b128 loads, convert, ds_store (128x32 -> bf16)
#pragma unroll
    for (int u = 0; u < 4; ++u) {
      int flat = u * 1024 + tid * 4;
      int rr = flat >> 5, kk = flat & 31;
      int row = row0 + rr; if (row >= M) row = M - 1;
      const float4 f = *(const float4*)(A + (size_t)row * K + kb + kk);
      bf16x4 pk = { (__bf16)f.x, (__bf16)f.y, (__bf16)f.z, (__bf16)f.w };
      *(bf16x4*)&sm.a[rr * 32 + kk] = pk;
    }
    asm volatile("s_wait_asynccnt 0x0" ::: "memory");
    __syncthreads();
    // ---- transpose+convert B into [n][k] bf16 (k contiguous)
#pragma unroll
    for (int u = 0; u < 4; ++u) {
      int p  = u * 256 + tid;
      int nn = p >> 4, kp = (p & 15) << 1;
      __bf16 b0 = (__bf16)sm.braw[kp * 64 + nn];
      __bf16 b1 = (__bf16)sm.braw[(kp + 1) * 64 + nn];
      sm.bt[nn * 32 + kp]     = b0;
      sm.bt[nn * 32 + kp + 1] = b1;
    }
    __syncthreads();
    // ---- fragments (2x ds_load_b128 each) + 4 WMMAs
    const int arow = wave * 16 + lm;
    bf16x8 alo = *(const bf16x8*)&sm.a[arow * 32 + khi * 8];
    bf16x8 ahi = *(const bf16x8*)&sm.a[arow * 32 + 16 + khi * 8];
    v16bf afrag = __builtin_shufflevector(alo, ahi,
                    0,1,2,3,4,5,6,7,8,9,10,11,12,13,14,15);
#pragma unroll
    for (int j = 0; j < 4; ++j) {
      const int nn = j * 16 + lm;
      bf16x8 blo = *(const bf16x8*)&sm.bt[nn * 32 + khi * 8];
      bf16x8 bhi = *(const bf16x8*)&sm.bt[nn * 32 + 16 + khi * 8];
      v16bf bfrag = __builtin_shufflevector(blo, bhi,
                      0,1,2,3,4,5,6,7,8,9,10,11,12,13,14,15);
      acc[j] = __builtin_amdgcn_wmma_f32_16x16x32_bf16(false, afrag, false, bfrag,
                                                       (short)0, acc[j], false, false);
    }
    __syncthreads();
  }

#pragma unroll
  for (int j = 0; j < 4; ++j) {
    const int col = col0 + j * 16 + lm;
    const float bval = bias ? bias[col] : 0.0f;
#pragma unroll
    for (int r = 0; r < 8; ++r) {
      int row = row0 + wave * 16 + r + 8 * khi;   // C layout: VGPR r -> M=r+8*khi
      if (row < M) C[(size_t)row * NC + col] = acc[j][r] + bval;
    }
  }
}

// --------------------------------------------- per-column mean/var over M rows
__global__ void colstats_kernel(const float* __restrict__ X, int M, int C,
                                float* mean, float* var) {
  const int c = blockIdx.x;
  __shared__ float ssum[256], ssq[256];
  float s = 0.0f, q = 0.0f;
  for (int r = threadIdx.x; r < M; r += blockDim.x) {
    float x = X[(size_t)r * C + c];
    s += x; q += x * x;
  }
  ssum[threadIdx.x] = s; ssq[threadIdx.x] = q;
  __syncthreads();
  for (int off = 128; off > 0; off >>= 1) {
    if ((int)threadIdx.x < off) {
      ssum[threadIdx.x] += ssum[threadIdx.x + off];
      ssq[threadIdx.x]  += ssq[threadIdx.x + off];
    }
    __syncthreads();
  }
  if (threadIdx.x == 0) {
    float m = ssum[0] / (float)M;
    mean[c] = m;
    var[c]  = ssq[0] / (float)M - m * m;
  }
}

__global__ void bn_apply_kernel(float* h, const float* mean, const float* var,
                                const float* g, const float* b, int M, int C, int relu) {
  int i = blockIdx.x * blockDim.x + threadIdx.x;
  if (i >= M * C) return;
  int c = i % C;
  float y = (h[i] - mean[c]) * rsqrtf(var[c] + EPSV) * g[c] + b[c];
  if (relu) y = fmaxf(y, 0.0f);
  h[i] = y;
}

// ------------------------------------------------- input BN + projection to HID
__global__ void bn_proj_kernel(const float* __restrict__ x, const float* g, const float* b,
                               const float* mean, const float* var,
                               const float* __restrict__ W, const float* bias,
                               float* __restrict__ h) {
  int i = blockIdx.x * blockDim.x + threadIdx.x;
  if (i >= N_NODES * HID) return;
  int n = i / HID, c = i % HID;
  float acc = bias[c];
#pragma unroll
  for (int f = 0; f < F_INP; ++f) {
    float xn = (x[n * F_INP + f] - mean[f]) * rsqrtf(var[f] + EPSV) * g[f] + b[f];
    acc += xn * W[f * HID + c];
  }
  h[i] = acc;
}

// ------------------------------------------------- edge attention: raw scores + seg-max
__global__ void edge_alpha_kernel(const float* __restrict__ q, const float* __restrict__ k,
                                  const float* __restrict__ ea, const float* __restrict__ We,
                                  const int* __restrict__ src, const int* __restrict__ dst,
                                  float* __restrict__ alpha, float* __restrict__ mbuf) {
  int i = blockIdx.x * blockDim.x + threadIdx.x;
  if (i >= N_EDGES * HEADS) return;
  int e = i / HEADS, hh = i % HEADS;
  int s = src[e], d = dst[e];
  float e0 = ea[e*EDIM+0], e1 = ea[e*EDIM+1], e2 = ea[e*EDIM+2], e3 = ea[e*EDIM+3];
  const float* qp = q + (size_t)d * QKV + hh * HID;
  const float* kp = k + (size_t)s * QKV + hh * HID;
  const float* wp = We + hh * HID;
  __builtin_prefetch(qp, 0, 1);
  __builtin_prefetch(kp, 0, 1);
  float acc = 0.0f;
  for (int c = 0; c < HID; ++c) {
    float ev = e0*wp[c] + e1*wp[QKV+c] + e2*wp[2*QKV+c] + e3*wp[3*QKV+c];
    acc += qp[c] * (kp[c] + ev);
  }
  acc *= 0.08838834764831845f;   // 1/sqrt(128)
  alpha[i] = acc;
  atomicMaxFloat(&mbuf[d * HEADS + hh], acc);
}

__global__ void edge_expsum_kernel(const int* __restrict__ dst, float* __restrict__ alpha,
                                   const float* __restrict__ mbuf, float* __restrict__ ssum) {
  int i = blockIdx.x * blockDim.x + threadIdx.x;
  if (i >= N_EDGES * HEADS) return;
  int e = i / HEADS, hh = i % HEADS;
  int d = dst[e];
  float a = __expf(alpha[i] - mbuf[d * HEADS + hh]);
  alpha[i] = a;
  atomicAdd(&ssum[d * HEADS + hh], a);
}

// one thread per (edge, channel): loop heads, single atomicAdd -> head-mean folded in
__global__ void edge_agg_kernel(const float* __restrict__ vmat, const float* __restrict__ ea,
                                const float* __restrict__ We,
                                const int* __restrict__ src, const int* __restrict__ dst,
                                const float* __restrict__ alpha, const float* __restrict__ ssum,
                                float* __restrict__ agg) {
  int i = blockIdx.x * blockDim.x + threadIdx.x;
  if (i >= N_EDGES * HID) return;
  int e = i / HID, c = i % HID;
  int s = src[e], d = dst[e];
  float e0 = ea[e*EDIM+0], e1 = ea[e*EDIM+1], e2 = ea[e*EDIM+2], e3 = ea[e*EDIM+3];
  __builtin_prefetch(&vmat[(size_t)s * QKV + c], 0, 1);
  float out = 0.0f;
#pragma unroll
  for (int hh = 0; hh < HEADS; ++hh) {
    float w = alpha[e * HEADS + hh] / (ssum[d * HEADS + hh] + 1e-16f);
    int cc = hh * HID + c;
    float ev = e0*We[cc] + e1*We[QKV+cc] + e2*We[2*QKV+cc] + e3*We[3*QKV+cc];
    out += w * (vmat[(size_t)s * QKV + cc] + ev);
  }
  atomicAdd(&agg[(size_t)d * HID + c], out * (1.0f / HEADS));
}

// ------------------------------------------------- attention pooling
__global__ void pool_score_kernel(const float* __restrict__ h,
                                  const float* __restrict__ W1, const float* __restrict__ b1,
                                  const float* __restrict__ W2, const float* __restrict__ b2,
                                  float* __restrict__ s, float* smax) {
  int n = blockIdx.x * blockDim.x + threadIdx.x;
  if (n >= N_NODES) return;
  float acc = b2[0];
  for (int j = 0; j < HID / 2; ++j) {
    float t = b1[j];
    for (int c = 0; c < HID; ++c) t += h[(size_t)n * HID + c] * W1[c * (HID / 2) + j];
    acc += tanhf(t) * W2[j];
  }
  s[n] = acc;
  atomicMaxFloat(smax, acc);
}

__global__ void pool_exp_kernel(float* s, const float* smax, float* wsum) {
  int n = blockIdx.x * blockDim.x + threadIdx.x;
  if (n >= N_NODES) return;
  float w = __expf(s[n] - smax[0]);
  s[n] = w;
  atomicAdd(wsum, w);
}

__global__ void pool_g_kernel(const float* __restrict__ h, const float* __restrict__ s,
                              const float* wsum, float* g) {
  const int c = blockIdx.x;
  __shared__ float red[256];
  float acc = 0.0f;
  for (int n = threadIdx.x; n < N_NODES; n += blockDim.x)
    acc += h[(size_t)n * HID + c] * s[n];
  red[threadIdx.x] = acc;
  __syncthreads();
  for (int off = 128; off > 0; off >>= 1) {
    if ((int)threadIdx.x < off) red[threadIdx.x] += red[threadIdx.x + off];
    __syncthreads();
  }
  if (threadIdx.x == 0) g[c] = red[0] / wsum[0];
}

__global__ void build_xg_kernel(const float* __restrict__ h, const float* __restrict__ g,
                                float* __restrict__ xg) {
  int i = blockIdx.x * blockDim.x + threadIdx.x;
  if (i >= N_NODES * 2 * HID) return;
  int n = i >> 8, c = i & 255;
  xg[i] = (c < HID) ? h[(size_t)n * HID + c] : g[c - HID];
}

__global__ void head_out_kernel(const float* __restrict__ y, const float* __restrict__ W3,
                                const float* b3, float* __restrict__ out, int C, int do_sig) {
  int n = blockIdx.x * blockDim.x + threadIdx.x;
  if (n >= N_NODES) return;
  float acc = b3[0];
  for (int j = 0; j < C; ++j) acc += y[(size_t)n * C + j] * W3[j];
  if (do_sig) acc = 1.0f / (1.0f + __expf(-acc));
  out[n] = acc;
}

// ---------------------------------------------------------------- launcher
static inline int ceil_div(int a, int b) { return (a + b - 1) / b; }

extern "C" void kernel_launch(void* const* d_in, const int* in_sizes, int n_in,
                              void* d_out, int out_size, void* d_ws, size_t ws_size,
                              hipStream_t stream) {
  const float* x       = (const float*)d_in[0];
  const int*   ei      = (const int*)  d_in[1];
  const float* ea      = (const float*)d_in[2];
  const float* in_g    = (const float*)d_in[3];
  const float* in_b    = (const float*)d_in[4];
  const float* proj_W  = (const float*)d_in[5];
  const float* proj_b  = (const float*)d_in[6];
  const float* Wq      = (const float*)d_in[7];
  const float* bq      = (const float*)d_in[8];
  const float* Wk      = (const float*)d_in[9];
  const float* bk      = (const float*)d_in[10];
  const float* Wv      = (const float*)d_in[11];
  const float* bv      = (const float*)d_in[12];
  const float* We      = (const float*)d_in[13];
  const float* Ws      = (const float*)d_in[14];
  const float* bs      = (const float*)d_in[15];
  const float* bn_g    = (const float*)d_in[16];
  const float* bn_b    = (const float*)d_in[17];
  const float* pool_W1 = (const float*)d_in[18];
  const float* pool_b1 = (const float*)d_in[19];
  const float* pool_W2 = (const float*)d_in[20];
  const float* pool_b2 = (const float*)d_in[21];
  const float* ord_W1  = (const float*)d_in[22];
  const float* ord_b1  = (const float*)d_in[23];
  const float* ord_g1  = (const float*)d_in[24];
  const float* ord_be1 = (const float*)d_in[25];
  const float* ord_W2  = (const float*)d_in[26];
  const float* ord_b2  = (const float*)d_in[27];
  const float* ord_g2  = (const float*)d_in[28];
  const float* ord_be2 = (const float*)d_in[29];
  const float* ord_W3  = (const float*)d_in[30];
  const float* ord_b3  = (const float*)d_in[31];
  const float* tim_W1  = (const float*)d_in[32];
  const float* tim_b1  = (const float*)d_in[33];
  const float* tim_g1  = (const float*)d_in[34];
  const float* tim_be1 = (const float*)d_in[35];
  const float* tim_W2  = (const float*)d_in[36];
  const float* tim_b2  = (const float*)d_in[37];
  const float* tim_g2  = (const float*)d_in[38];
  const float* tim_be2 = (const float*)d_in[39];
  const float* tim_W3  = (const float*)d_in[40];
  const float* tim_b3  = (const float*)d_in[41];
  const float* pri_W1  = (const float*)d_in[42];
  const float* pri_b1  = (const float*)d_in[43];
  const float* pri_W2  = (const float*)d_in[44];
  const float* pri_b2  = (const float*)d_in[45];

  const int* src = ei;
  const int* dst = ei + N_EDGES;
  float* out = (float*)d_out;

  // ------------ workspace carve-up (floats; every block 16B-aligned)
  float* w = (float*)d_ws;
  size_t o = 0;
  float* stat_m = w + o; o += F_INP;                    // 8
  float* stat_v = w + o; o += F_INP;                    // 8
  float* cm     = w + o; o += HID;
  float* cv     = w + o; o += HID;
  float* smax   = w + o; o += 4;                        // padded scalars
  float* wsum   = w + o; o += 4;
  float* gvec   = w + o; o += HID;
  float* h      = w + o; o += (size_t)N_NODES * HID;
  float* hs     = w + o; o += (size_t)N_NODES * HID;
  float* qb     = w + o; o += (size_t)N_NODES * QKV;
  float* kb     = w + o; o += (size_t)N_NODES * QKV;
  float* vb     = w + o; o += (size_t)N_NODES * QKV;
  float* alpha  = w + o; o += (size_t)N_EDGES * HEADS;
  float* mbuf   = w + o; o += (size_t)N_NODES * HEADS;
  float* sbuf   = w + o; o += (size_t)N_NODES * HEADS;
  float* agg    = w + o; o += (size_t)N_NODES * HID;
  float* pscore = w + o; o += N_NODES;
  o = (o + 3) & ~(size_t)3;
  float* xg     = w + o; o += (size_t)N_NODES * 2 * HID;
  float* y1     = w + o; o += (size_t)N_NODES * HID;
  float* y2     = w + o; o += (size_t)N_NODES * (HID / 2);

  const int TPB = 256;
  const int NH  = N_NODES * HID;
  const int GM  = ceil_div(N_NODES, 128);   // 79 row-tiles

  // ------------ input BN + projection
  colstats_kernel<<<F_INP, 256, 0, stream>>>(x, N_NODES, F_INP, stat_m, stat_v);
  bn_proj_kernel<<<ceil_div(NH, TPB), TPB, 0, stream>>>(x, in_g, in_b, stat_m, stat_v,
                                                        proj_W, proj_b, h);

  // ------------ transformer layers
  for (int i = 0; i < LAYERS; ++i) {
    if (i > 0)   // res == h at layer entry -> h + res == 2h
      scale2_kernel<<<ceil_div(NH, TPB), TPB, 0, stream>>>(h, NH);

    const float* Wq_i = Wq + (size_t)i * HID * QKV;
    const float* Wk_i = Wk + (size_t)i * HID * QKV;
    const float* Wv_i = Wv + (size_t)i * HID * QKV;
    const float* We_i = We + (size_t)i * EDIM * QKV;
    const float* Ws_i = Ws + (size_t)i * HID * HID;

    wmma_gemm_bias<<<dim3(GM, QKV / 64), 256, 0, stream>>>(h, Wq_i, bq + (size_t)i * QKV, qb, N_NODES, HID, QKV);
    wmma_gemm_bias<<<dim3(GM, QKV / 64), 256, 0, stream>>>(h, Wk_i, bk + (size_t)i * QKV, kb, N_NODES, HID, QKV);
    wmma_gemm_bias<<<dim3(GM, QKV / 64), 256, 0, stream>>>(h, Wv_i, bv + (size_t)i * QKV, vb, N_NODES, HID, QKV);
    wmma_gemm_bias<<<dim3(GM, HID / 64), 256, 0, stream>>>(h, Ws_i, bs + (size_t)i * HID, hs, N_NODES, HID, HID);

    fill_kernel<<<ceil_div(N_NODES * HEADS, TPB), TPB, 0, stream>>>(mbuf, -__builtin_inff(), N_NODES * HEADS);
    fill_kernel<<<ceil_div(N_NODES * HEADS, TPB), TPB, 0, stream>>>(sbuf, 0.0f, N_NODES * HEADS);
    fill_kernel<<<ceil_div(NH, TPB), TPB, 0, stream>>>(agg, 0.0f, NH);

    edge_alpha_kernel<<<ceil_div(N_EDGES * HEADS, TPB), TPB, 0, stream>>>(qb, kb, ea, We_i, src, dst, alpha, mbuf);
    edge_expsum_kernel<<<ceil_div(N_EDGES * HEADS, TPB), TPB, 0, stream>>>(dst, alpha, mbuf, sbuf);
    edge_agg_kernel<<<ceil_div(N_EDGES * HID, TPB), TPB, 0, stream>>>(vb, ea, We_i, src, dst, alpha, sbuf, agg);

    add2_kernel<<<ceil_div(NH, TPB), TPB, 0, stream>>>(h, agg, hs, NH);
    colstats_kernel<<<HID, 256, 0, stream>>>(h, N_NODES, HID, cm, cv);
    bn_apply_kernel<<<ceil_div(NH, TPB), TPB, 0, stream>>>(h, cm, cv, bn_g + (size_t)i * HID,
                                                           bn_b + (size_t)i * HID, N_NODES, HID,
                                                           (i < LAYERS - 1) ? 1 : 0);
  }

  // ------------ attention pooling
  fill_kernel<<<1, 32, 0, stream>>>(smax, -__builtin_inff(), 1);
  fill_kernel<<<1, 32, 0, stream>>>(wsum, 0.0f, 1);
  pool_score_kernel<<<ceil_div(N_NODES, TPB), TPB, 0, stream>>>(h, pool_W1, pool_b1, pool_W2, pool_b2, pscore, smax);
  pool_exp_kernel<<<ceil_div(N_NODES, TPB), TPB, 0, stream>>>(pscore, smax, wsum);
  pool_g_kernel<<<HID, 256, 0, stream>>>(h, pscore, wsum, gvec);
  build_xg_kernel<<<ceil_div(N_NODES * 2 * HID, TPB), TPB, 0, stream>>>(h, gvec, xg);

  // ------------ order head
  wmma_gemm_bias<<<dim3(GM, HID / 64), 256, 0, stream>>>(xg, ord_W1, ord_b1, y1, N_NODES, 2 * HID, HID);
  colstats_kernel<<<HID, 256, 0, stream>>>(y1, N_NODES, HID, cm, cv);
  bn_apply_kernel<<<ceil_div(NH, TPB), TPB, 0, stream>>>(y1, cm, cv, ord_g1, ord_be1, N_NODES, HID, 1);
  wmma_gemm_bias<<<dim3(GM, 1), 256, 0, stream>>>(y1, ord_W2, ord_b2, y2, N_NODES, HID, HID / 2);
  colstats_kernel<<<HID / 2, 256, 0, stream>>>(y2, N_NODES, HID / 2, cm, cv);
  bn_apply_kernel<<<ceil_div(N_NODES * (HID / 2), TPB), TPB, 0, stream>>>(y2, cm, cv, ord_g2, ord_be2, N_NODES, HID / 2, 1);
  head_out_kernel<<<ceil_div(N_NODES, TPB), TPB, 0, stream>>>(y2, ord_W3, ord_b3, out + 0, HID / 2, 0);

  // ------------ time head
  wmma_gemm_bias<<<dim3(GM, HID / 64), 256, 0, stream>>>(xg, tim_W1, tim_b1, y1, N_NODES, 2 * HID, HID);
  colstats_kernel<<<HID, 256, 0, stream>>>(y1, N_NODES, HID, cm, cv);
  bn_apply_kernel<<<ceil_div(NH, TPB), TPB, 0, stream>>>(y1, cm, cv, tim_g1, tim_be1, N_NODES, HID, 1);
  wmma_gemm_bias<<<dim3(GM, 1), 256, 0, stream>>>(y1, tim_W2, tim_b2, y2, N_NODES, HID, HID / 2);
  colstats_kernel<<<HID / 2, 256, 0, stream>>>(y2, N_NODES, HID / 2, cm, cv);
  bn_apply_kernel<<<ceil_div(N_NODES * (HID / 2), TPB), TPB, 0, stream>>>(y2, cm, cv, tim_g2, tim_be2, N_NODES, HID / 2, 1);
  head_out_kernel<<<ceil_div(N_NODES, TPB), TPB, 0, stream>>>(y2, tim_W3, tim_b3, out + N_NODES, HID / 2, 0);

  // ------------ priority head (no BN, relu then sigmoid)
  wmma_gemm_bias<<<dim3(GM, 1), 256, 0, stream>>>(xg, pri_W1, pri_b1, y2, N_NODES, 2 * HID, HID / 2);
  relu_kernel<<<ceil_div(N_NODES * (HID / 2), TPB), TPB, 0, stream>>>(y2, N_NODES * (HID / 2));
  head_out_kernel<<<ceil_div(N_NODES, TPB), TPB, 0, stream>>>(y2, pri_W2, pri_b2, out + 2 * N_NODES, HID / 2, 1);
}
